// NeuralODE_88991722373458
// MI455X (gfx1250) — matrix operand order repaired and checked by reading, not compiled, not run on hardware
//
#include <hip/hip_runtime.h>

#define B_ROWS   65536
#define T_STEPS  20
#define N_SUB    4

typedef __attribute__((ext_vector_type(16))) __bf16 v16bf;
typedef __attribute__((ext_vector_type(8)))  __bf16 v8bf;
typedef __attribute__((ext_vector_type(8)))  float  v8f;
typedef __attribute__((ext_vector_type(4)))  float  v4f;
typedef __bf16 bf16_t;

// ---------------- LDS layout (bytes) ----------------
// Weight fragments: 32 lanes * 16 bf16 = 1024B per 32x16 fragment.
static constexpr int OFF_W1   = 0;                  // 16 frags (K=64 /32 * N=128 /16)
static constexpr int OFF_W2   = OFF_W1 + 16 * 1024; // 64 frags
static constexpr int OFF_W3   = OFF_W2 + 64 * 1024; // 32 frags
static constexpr int OFF_BIAS = OFF_W3 + 32 * 1024; // b1(128)+b2(256)+b3(64)+Win(64)+bin(64)+Wout(64) f32
static constexpr int OFF_WAVE = OFF_BIAS + 640 * 4; // per-wave scratch follows
// Per-wave scratch:
//   k buffers : 6 * [2 frag][32 lane][20 f32 (16 used, staggered)] = 6*5120B
//   h1 (bf16) : [4 frag][32 lane][24 bf16 (16 used, staggered)] + 16 pad = 6176B
//   h2 (bf16) : [8 frag][32 lane][24 bf16] + 16 pad                      = 12320B
static constexpr int KBUF_F     = 1280;              // floats per k buffer
static constexpr int OFF_H1     = 6 * 5120;          // within wave scratch
static constexpr int OFF_H2     = OFF_H1 + 6176;
static constexpr int WAVE_BYTES = OFF_H2 + 12320;    // 49216
static constexpr int SMEM_TOTAL = OFF_WAVE + 4 * WAVE_BYTES; // 314112 < 320KB

// dopri5 tableau (lower-triangular A rows for stages 2..6)
__device__ __constant__ float AC[5][5] = {
  {1.f/5.f, 0.f, 0.f, 0.f, 0.f},
  {3.f/40.f, 9.f/40.f, 0.f, 0.f, 0.f},
  {44.f/45.f, -56.f/15.f, 32.f/9.f, 0.f, 0.f},
  {19372.f/6561.f, -25360.f/2187.f, 64448.f/6561.f, -212.f/729.f, 0.f},
  {9017.f/3168.f, -355.f/33.f, 46732.f/5247.f, 49.f/176.f, -5103.f/18656.f}};

__device__ __forceinline__ v16bf ld_frag(const bf16_t* p) {
  v8bf a = *(const v8bf*)(p);
  v8bf b = *(const v8bf*)(p + 8);
  v16bf w;
#pragma unroll
  for (int i = 0; i < 8; ++i) { w[i] = a[i]; w[8 + i] = b[i]; }
  return w;
}

__device__ __forceinline__ v8f wmma_bf16(v16bf a, v16bf b, v8f c) {
  return __builtin_amdgcn_wmma_f32_16x16x32_bf16(false, a, false, b, (short)0, c, false, false);
}

__device__ __forceinline__ float elu(float v) {
  return v > 0.f ? v : (__expf(v) - 1.f);
}

__global__ __launch_bounds__(128, 1)
void node_dopri5_wmma(const float* __restrict__ x,
                      const float* __restrict__ t_span,
                      const float* __restrict__ W_in,
                      const float* __restrict__ b_in,
                      const float* __restrict__ W1g,
                      const float* __restrict__ b1g,
                      const float* __restrict__ W2g,
                      const float* __restrict__ b2g,
                      const float* __restrict__ W3g,
                      const float* __restrict__ b3g,
                      const float* __restrict__ W_out,
                      const float* __restrict__ b_out,
                      float* __restrict__ out) {
  extern __shared__ char smem[];
  const int tid  = threadIdx.x;
  const int lane = tid & 31;
  const int wave = tid >> 5;
  const int hi   = lane >> 4;   // which 16-lane half
  const int col  = lane & 15;

  bf16_t* w1s = (bf16_t*)(smem + OFF_W1);
  bf16_t* w2s = (bf16_t*)(smem + OFF_W2);
  bf16_t* w3s = (bf16_t*)(smem + OFF_W3);
  float* b1S   = (float*)(smem + OFF_BIAS);
  float* b2S   = b1S + 128;
  float* b3S   = b2S + 256;
  float* WinS  = b3S + 64;
  float* binS  = WinS + 64;
  float* WoutS = binS + 64;

  // ---- build bf16 weight fragments in B-operand layout ----
  // Fragment f = kt*NT + nt ; element (lane l, slot i): k = kt*32 + (i>>3)*16 + (l>>4)*8 + (i&7),
  // n = nt*16 + (l&15).  Each lane's 16 bf16 are contiguous (32B).
  auto build = [&](bf16_t* dst, const float* src, int K, int N) {
    const int NT = N >> 4;
    const int total = K * N;
    for (int idx = tid; idx < total; idx += 128) {
      int f   = idx >> 9;
      int rem = idx & 511;
      int l   = rem >> 4;
      int i   = rem & 15;
      int nt  = f % NT;
      int kt  = f / NT;
      int k   = kt * 32 + ((i >> 3) << 4) + ((l >> 4) << 3) + (i & 7);
      int n   = nt * 16 + (l & 15);
      dst[idx] = (bf16_t)src[k * N + n];
    }
  };
  build(w1s, W1g, 64, 128);
  build(w2s, W2g, 128, 256);
  build(w3s, W3g, 256, 64);
  if (tid < 128) b1S[tid] = b1g[tid];
  for (int i = tid; i < 256; i += 128) b2S[i] = b2g[i];
  if (tid < 64) {
    b3S[tid]  = b3g[tid];
    WinS[tid] = W_in[tid];
    binS[tid] = b_in[tid];
    WoutS[tid] = W_out[tid];
  }
  __syncthreads();

  // ---- per-wave scratch ----
  char*  wbase = smem + OFF_WAVE + wave * WAVE_BYTES;
  float* kbufs = (float*)wbase;
  bf16_t* h1s  = (bf16_t*)(wbase + OFF_H1);
  bf16_t* h2s  = (bf16_t*)(wbase + OFF_H2);

  const int rowBase = blockIdx.x * 64 + wave * 16;
  const int myRow   = rowBase + col;   // A-layout: every element of this lane has M = col
  const float xr    = x[myRow];
  const float bo    = b_out[0];

  // A-fragment element -> K index map for (lane, slot i), i in [0,32) over 2 frags
#define KOF(i) ( ((i) >> 4) * 32 + ((((i) & 15) >> 3) << 4) + (hi << 3) + ((i) & 7) )

  float y[32];
#pragma unroll
  for (int i = 0; i < 32; ++i) { int k = KOF(i); y[i] = xr * WinS[k] + binS[k]; }

  auto writeOut = [&](int t) {
    float s = 0.f;
#pragma unroll
    for (int i = 0; i < 32; ++i) { int k = KOF(i); s += y[i] * WoutS[k]; }
    s += __shfl_xor(s, 16, 32);          // combine the two K-halves of row M=col
    if (lane < 16) out[t * B_ROWS + myRow] = s + bo;
  };

  // f(arg) -> kbufs[kidx] ; arg is per-lane f32 in A-fragment order
  auto fEval = [&](const float (&arg)[32], int kidx) {
    v16bf aF0, aF1;
#pragma unroll
    for (int i = 0; i < 16; ++i) { aF0[i] = (bf16_t)arg[i]; aF1[i] = (bf16_t)arg[16 + i]; }

    // ---- layer 1: [16,64] x [64,128] ----
#pragma unroll
    for (int nt = 0; nt < 8; ++nt) {
      float bv = b1S[nt * 16 + col];
      v8f acc = {bv, bv, bv, bv, bv, bv, bv, bv};
      acc = wmma_bf16(aF0, ld_frag(w1s + (0 * 8 + nt) * 512 + lane * 16), acc);
      acc = wmma_bf16(aF1, ld_frag(w1s + (1 * 8 + nt) * 512 + lane * 16), acc);
      const int n  = nt * 16 + col;
      const int fr = n >> 5, rem = n & 31;
      const int hf = (rem >> 3) & 1;
      const int ii = ((rem >> 4) << 3) + (rem & 7);
      bf16_t* dst = h1s + (fr * 32 + 16 * hf) * 24 + hf * 8 + ii;
#pragma unroll
      for (int r = 0; r < 8; ++r) {
        int m = r + 8 * hi;                        // D layout row
        dst[m * 24] = (bf16_t)elu(acc[r]);         // scatter into A-order
      }
    }
    asm volatile("s_wait_dscnt 0x0" ::: "memory");
    v16bf a2[4];
#pragma unroll
    for (int f = 0; f < 4; ++f) a2[f] = ld_frag(h1s + (f * 32 + lane) * 24 + hi * 8);

    // ---- layer 2: [16,128] x [128,256] ----
#pragma unroll
    for (int nt = 0; nt < 16; ++nt) {
      float bv = b2S[nt * 16 + col];
      v8f acc = {bv, bv, bv, bv, bv, bv, bv, bv};
#pragma unroll
      for (int kt = 0; kt < 4; ++kt)
        acc = wmma_bf16(a2[kt], ld_frag(w2s + (kt * 16 + nt) * 512 + lane * 16), acc);
      const int n  = nt * 16 + col;
      const int fr = n >> 5, rem = n & 31;
      const int hf = (rem >> 3) & 1;
      const int ii = ((rem >> 4) << 3) + (rem & 7);
      bf16_t* dst = h2s + (fr * 32 + 16 * hf) * 24 + hf * 8 + ii;
#pragma unroll
      for (int r = 0; r < 8; ++r) {
        int m = r + 8 * hi;
        dst[m * 24] = (bf16_t)elu(acc[r]);
      }
    }
    asm volatile("s_wait_dscnt 0x0" ::: "memory");
    v16bf a3[8];
#pragma unroll
    for (int f = 0; f < 8; ++f) a3[f] = ld_frag(h2s + (f * 32 + lane) * 24 + hi * 8);

    // ---- layer 3: [16,256] x [256,64] -> k buffer (f32, A-order) ----
    float* kp = kbufs + kidx * KBUF_F;
#pragma unroll
    for (int nt = 0; nt < 4; ++nt) {
      float bv = b3S[nt * 16 + col];
      v8f acc = {bv, bv, bv, bv, bv, bv, bv, bv};
#pragma unroll
      for (int kt = 0; kt < 8; ++kt)
        acc = wmma_bf16(a3[kt], ld_frag(w3s + (kt * 4 + nt) * 512 + lane * 16), acc);
      const int n  = nt * 16 + col;
      const int fr = n >> 5, rem = n & 31;
      const int hf = (rem >> 3) & 1;
      const int ii = ((rem >> 4) << 3) + (rem & 7);
      float* dstf = kp + (fr * 32 + 16 * hf) * 20 + hf * 4 + ii;
#pragma unroll
      for (int r = 0; r < 8; ++r) dstf[(r + 8 * hi) * 20] = acc[r];
    }
    asm volatile("s_wait_dscnt 0x0" ::: "memory");
  };

  // tmp[i] += c * k_j[i]  (per-lane contiguous 2x64B chunks, 16B aligned)
  auto accumK = [&](int kidx, float c, float (&t32)[32]) {
    const float* kp = kbufs + kidx * KBUF_F;
#pragma unroll
    for (int f = 0; f < 2; ++f) {
      const v4f* p = (const v4f*)(kp + (f * 32 + lane) * 20 + hi * 4);
#pragma unroll
      for (int q = 0; q < 4; ++q) {
        v4f v = p[q];
#pragma unroll
        for (int e = 0; e < 4; ++e) t32[f * 16 + q * 4 + e] += c * v[e];
      }
    }
  };

  writeOut(0);

  for (int t = 1; t < T_STEPS; ++t) {
    const float dt = (t_span[t] - t_span[t - 1]) * (1.0f / N_SUB);
    for (int ss = 0; ss < N_SUB; ++ss) {
      for (int s = 0; s < 6; ++s) {
        float tmp[32];
#pragma unroll
        for (int i = 0; i < 32; ++i) tmp[i] = y[i];
        for (int j = 0; j < s; ++j) accumK(j, dt * AC[s - 1][j], tmp);
        fEval(tmp, s);
      }
      constexpr float BC[6] = {35.f/384.f, 0.f, 500.f/1113.f,
                               125.f/192.f, -2187.f/6784.f, 11.f/84.f};
#pragma unroll
      for (int j = 0; j < 6; ++j) {
        if (j == 1) continue;                // B2 == 0
        accumK(j, dt * BC[j], y);
      }
    }
    writeOut(t);
  }
#undef KOF
}

extern "C" void kernel_launch(void* const* d_in, const int* in_sizes, int n_in,
                              void* d_out, int out_size, void* d_ws, size_t ws_size,
                              hipStream_t stream) {
  (void)in_sizes; (void)n_in; (void)out_size; (void)d_ws; (void)ws_size;
  const float* x      = (const float*)d_in[0];
  const float* t_span = (const float*)d_in[1];
  const float* W_in   = (const float*)d_in[2];
  const float* b_in   = (const float*)d_in[3];
  const float* W1     = (const float*)d_in[4];
  const float* b1     = (const float*)d_in[5];
  const float* W2     = (const float*)d_in[6];
  const float* b2     = (const float*)d_in[7];
  const float* W3     = (const float*)d_in[8];
  const float* b3     = (const float*)d_in[9];
  const float* W_out  = (const float*)d_in[10];
  const float* b_out  = (const float*)d_in[11];
  float* out = (float*)d_out;

  // allow >64KB dynamic LDS if the runtime gates it (no-op / ignored otherwise)
  (void)hipFuncSetAttribute((const void*)node_dopri5_wmma,
                            hipFuncAttributeMaxDynamicSharedMemorySize, SMEM_TOTAL);

  dim3 grid(B_ROWS / 64);
  dim3 block(128);
  node_dopri5_wmma<<<grid, block, SMEM_TOTAL, stream>>>(
      x, t_span, W_in, b_in, W1, b1, W2, b2, W3, b3, W_out, b_out, out);
}
